// BloomAttention_33904471834738
// MI455X (gfx1250) — matrix operand order repaired and checked by reading, not compiled, not run on hardware
//
#include <hip/hip_runtime.h>
#include <hip/hip_bf16.h>

// Bloom attention block, MI455X (gfx1250, wave32, WMMA bf16 + TDM).
// Pipeline: cvt(f32->bf16) -> QKV GEMM (WMMA) -> flash attention (WMMA, K tile
// staged by Tensor Data Mover) -> dense GEMM (WMMA).
// Workspace layout (bytes):
//   [0,            16777216)   hidden_states bf16  (2048x4096)
//   [16777216,    117440512)   W_qkv bf16          (4096x12288)
//   [117440512,   150994944)   W_dense bf16        (4096x4096)
//   [150994944,   201326592)   fused qkv bf16      (2048x12288)
//   [201326592,   218103808)   ctx bf16            (2048x4096)

typedef __bf16 bf16_t;
typedef __attribute__((ext_vector_type(8)))  bf16_t v8bf;
typedef __attribute__((ext_vector_type(16))) bf16_t v16bf;
typedef __attribute__((ext_vector_type(8)))  float  v8f;
typedef __attribute__((ext_vector_type(4)))  unsigned int u32x4;
typedef __attribute__((ext_vector_type(8)))  int i32x8;
typedef __attribute__((ext_vector_type(4)))  int i32x4;

#define NB   2
#define SS   1024
#define HH   4096
#define NHD  32
#define HDIM 128
#define MTOK 2048
#define INV_NORM 0.08838834764831845f   // 1/sqrt(128)

__device__ __forceinline__ v16bf cat16(v8bf lo, v8bf hi) {
  v16bf r;
#pragma unroll
  for (int i = 0; i < 8; ++i) { r[i] = lo[i]; r[i + 8] = hi[i]; }
  return r;
}

__device__ __forceinline__ v8f wmma_bf16(v16bf a, v16bf b, v8f c) {
  // D = A(16x32 bf16) x B(32x16 bf16) + C(16x16 f32)
  return __builtin_amdgcn_wmma_f32_16x16x32_bf16(
      /*neg_a=*/false, a, /*neg_b=*/false, b,
      /*c_mod=*/(short)0, c, /*reuse_a=*/false, /*reuse_b=*/false);
}

// TDM: load a 64-row x 128-elt bf16 tile (row stride 12288 elts) into LDS,
// padding 4 DWORDs after every row (64 DWORDs) -> LDS row stride 136 elts.
// D# per CDNA5 ISA ch.8: group0 {count/lds_addr/global_addr/type=2},
// group1 {data_size=2B, pad_enable, pad_interval=5(64DW), pad_amount=3(4DW),
//         tensor_dim0=128, tensor_dim1=64, tile_dim0=128, tile_dim1=64,
//         tensor_dim0_stride=12288}.  (clang-23 6-arg builtin form.)
__device__ __forceinline__ void tdm_load_k_tile(const bf16_t* gptr, unsigned ldsAddr) {
  unsigned long long ga = (unsigned long long)gptr;
  u32x4 g0 = {0u, 0u, 0u, 0u};
  g0[0] = 1u;                                              // count=1 (valid)
  g0[1] = ldsAddr;                                         // lds_addr (bytes)
  g0[2] = (unsigned)(ga & 0xFFFFFFFFull);                  // global_addr[31:0]
  g0[3] = (unsigned)((ga >> 32) & 0x01FFFFFFull) | 0x80000000u;  // [56:32] | type=2
  i32x8 g1 = {0, 0, 0, 0, 0, 0, 0, 0};
  g1[0] = (1 << 16) | (1 << 20) | (5 << 22) | (3 << 25);   // 2B, pad 64DW/+4DW
  g1[1] = (int)(HDIM << 16);                               // tensor_dim0 = 128
  g1[2] = (int)(64u << 16);                                // tensor_dim1 = 64
  g1[3] = (int)(HDIM << 16);                               // tile_dim0 = 128
  g1[4] = 64;                                              // tile_dim1 = 64
  g1[5] = 3 * HH;                                          // dim0 stride = 12288
  i32x4 g2 = {0, 0, 0, 0};
  i32x4 g3 = {0, 0, 0, 0};
  i32x8 g4 = {0, 0, 0, 0, 0, 0, 0, 0};
  __builtin_amdgcn_tensor_load_to_lds(g0, g1, g2, g3, g4, 0);
}

// ---------------------------------------------------------------- cvt kernel
__global__ void bloom_cvt_f32_bf16(const float* __restrict__ in,
                                   bf16_t* __restrict__ out, long n) {
  long i = ((long)blockIdx.x * blockDim.x + threadIdx.x) * 4;
  long stride = (long)gridDim.x * blockDim.x * 4;
  for (long j = i; j < n; j += stride) {
    float4 v = *(const float4*)(in + j);
    out[j + 0] = (bf16_t)v.x;
    out[j + 1] = (bf16_t)v.y;
    out[j + 2] = (bf16_t)v.z;
    out[j + 3] = (bf16_t)v.w;
  }
}

// ------------------------------------------------------------- GEMM (WMMA)
// C[M,N] = A[M,K](bf16) * B[K,N](bf16) + bias; optional +residual, f32 out.
// Block tile 128x128, BK=32, 256 threads = 8 waves (2x4 wave grid, 64x32 each).
// Register double-buffer on the global->LDS staging; B fragments hoisted so
// LDS traffic is 12 ds_load_b128 per 8 WMMA per K-step.
template <bool OUT_F32>
__global__ __launch_bounds__(256)
void bloom_gemm_bf16(const bf16_t* __restrict__ A, const bf16_t* __restrict__ Bw,
                     const float* __restrict__ bias, const float* __restrict__ resid,
                     bf16_t* __restrict__ outb, float* __restrict__ outf,
                     int M, int N, int K) {
  __shared__ bf16_t sA[128 * 40];  // [row][k]  stride 40 elems (80B, pad)
  __shared__ bf16_t sB[128 * 40];  // [col][k]  stride 40 elems (transposed)

  const int tid = threadIdx.x;
  const int mBase = blockIdx.y * 128;
  const int nBase = blockIdx.x * 128;

  const int wave = tid >> 5;
  const int lane = tid & 31;
  const int half = lane >> 4;
  const int l    = lane & 15;
  const int wm = wave >> 2;   // 0..1 -> 64 rows each
  const int wn = wave & 3;    // 0..3 -> 32 cols each

  // cooperative load mapping
  const int ar = tid >> 1, ak = (tid & 1) * 16;   // A: 128 rows x 32 k
  const int bk = tid >> 3, bc = (tid & 7) * 16;   // B: 32 k x 128 cols

  const bf16_t* Aptr = A + (long)(mBase + ar) * K + ak;
  const bf16_t* Bptr = Bw + (long)bk * N + nBase + bc;

  v8f acc[4][2];
#pragma unroll
  for (int i = 0; i < 4; ++i)
#pragma unroll
    for (int j = 0; j < 2; ++j) acc[i][j] = {};

  // stage tile k0=0 into registers
  v8bf ra0 = *(const v8bf*)(Aptr);
  v8bf ra1 = *(const v8bf*)(Aptr + 8);
  v8bf rb0 = *(const v8bf*)(Bptr);
  v8bf rb1 = *(const v8bf*)(Bptr + 8);

  for (int k0 = 0; k0 < K; k0 += 32) {
    // ---- commit staged registers to LDS
    *(v8bf*)&sA[ar * 40 + ak]     = ra0;
    *(v8bf*)&sA[ar * 40 + ak + 8] = ra1;
#pragma unroll
    for (int i = 0; i < 8; ++i) {
      sB[(bc + i) * 40 + bk]     = rb0[i];
      sB[(bc + 8 + i) * 40 + bk] = rb1[i];
    }
    __syncthreads();

    // ---- issue next tile's global loads (overlap with WMMA below)
    if (k0 + 32 < K) {
      ra0 = *(const v8bf*)(Aptr + k0 + 32);
      ra1 = *(const v8bf*)(Aptr + k0 + 40);
      rb0 = *(const v8bf*)(Bptr + (long)(k0 + 32) * N);
      rb1 = *(const v8bf*)(Bptr + (long)(k0 + 32) * N + 8);
    }

    // ---- WMMA: hoist the 2 B fragments, stream 4 A fragments
    v16bf bfrag[2];
#pragma unroll
    for (int fn = 0; fn < 2; ++fn) {
      const int col = wn * 32 + fn * 16 + l;
      v8bf blo = *(v8bf*)&sB[col * 40 + half * 8];
      v8bf bhi = *(v8bf*)&sB[col * 40 + 16 + half * 8];
      bfrag[fn] = cat16(blo, bhi);
    }
#pragma unroll
    for (int fm = 0; fm < 4; ++fm) {
      const int row = wm * 64 + fm * 16 + l;
      v8bf alo = *(v8bf*)&sA[row * 40 + half * 8];
      v8bf ahi = *(v8bf*)&sA[row * 40 + 16 + half * 8];
      v16bf afrag = cat16(alo, ahi);
#pragma unroll
      for (int fn = 0; fn < 2; ++fn)
        acc[fm][fn] = wmma_bf16(afrag, bfrag[fn], acc[fm][fn]);
    }
    __syncthreads();
  }

  // ---- epilogue (C layout: VGPR r -> M=r(+8 for hi half), lane -> N)
#pragma unroll
  for (int fm = 0; fm < 4; ++fm) {
#pragma unroll
    for (int fn = 0; fn < 2; ++fn) {
      const int col = nBase + wn * 32 + fn * 16 + l;
      const float bv = bias[col];
#pragma unroll
      for (int r = 0; r < 8; ++r) {
        const long row = mBase + wm * 64 + fm * 16 + r + half * 8;
        float v = acc[fm][fn][r] + bv;
        if constexpr (OUT_F32) {
          outf[row * N + col] = v + resid[row * N + col];
        } else {
          outb[row * N + col] = (bf16_t)v;
        }
      }
    }
  }
}

// --------------------------------------------------- flash attention (WMMA)
// grid = (S/64 q-tiles, B*NH heads), 128 threads = 4 waves, 16 q-rows/wave.
// qkv layout per token row (12288): head h -> [h*384 + 0:128)=q, +128=k, +256=v
// K tile staged by the Tensor Data Mover (pad feature gives the 136-elt LDS
// stride); V staged manually (transposed).
__global__ __launch_bounds__(128)
void bloom_attn(const bf16_t* __restrict__ qkv, bf16_t* __restrict__ ctx) {
  __shared__ bf16_t sK[64 * 136];    // [krow][hd]   stride 136 (272B, TDM pad)
  __shared__ bf16_t sVT[128 * 72];   // [hd][krow]   stride 72  (144B, pad)
  __shared__ bf16_t sP[4][16 * 64];  // per-wave probs, row-major 16x64

  const int tid  = threadIdx.x;
  const int wave = tid >> 5;
  const int lane = tid & 31;
  const int half = lane >> 4;
  const int l    = lane & 15;

  const int qtile = blockIdx.x;          // 0..15
  const int bh    = blockIdx.y;          // 0..63
  const int b     = bh >> 5;
  const int h     = bh & 31;
  const float slope = exp2f(-0.25f * (float)(h + 1));
  const unsigned ldsK = (unsigned)(unsigned long long)&sK[0];  // LDS byte addr

  // Q A-fragments (kept in registers; 4 chunks of K=32 covering HD=128)
  const long qTok = (long)b * SS + qtile * 64 + wave * 16 + l;
  const bf16_t* qptr = qkv + qTok * 12288 + h * 384;
  v16bf qfrag[4];
#pragma unroll
  for (int kk = 0; kk < 4; ++kk) {
    v8bf lo = *(const v8bf*)(qptr + kk * 32 + half * 8);
    v8bf hi = *(const v8bf*)(qptr + kk * 32 + 16 + half * 8);
    qfrag[kk] = cat16(lo, hi);
  }

  float mstate[8], lstate[8];
  v8f ctxacc[8];
#pragma unroll
  for (int r = 0; r < 8; ++r) { mstate[r] = -3.0e38f; lstate[r] = 0.0f; }
#pragma unroll
  for (int t = 0; t < 8; ++t) ctxacc[t] = {};

  const int nkb = qtile + 1;  // causal: only k-blocks up to the q tile
  for (int kb = 0; kb < nkb; ++kb) {
    __syncthreads();
    // ---- TDM: DMA the K tile (64x128 bf16, padded) into sK (one wave issues)
    if (wave == 0)
      tdm_load_k_tile(qkv + ((long)b * SS + kb * 64) * 12288 + h * 384 + 128, ldsK);

    // ---- stage V (transposed) manually: 64 rows x 128 hd
    {
      const int r  = tid >> 1;
      const int hc = (tid & 1) * 64;
      const long kTok = (long)b * SS + kb * 64 + r;
      const bf16_t* vp = qkv + kTok * 12288 + h * 384 + 256 + hc;
#pragma unroll
      for (int c = 0; c < 8; ++c) {
        v8bf vv = *(const v8bf*)(vp + c * 8);
#pragma unroll
        for (int i = 0; i < 8; ++i) sVT[(hc + c * 8 + i) * 72 + r] = vv[i];
      }
      if (kb + 1 < nkb)  // prefetch next block's V row for this thread
        __builtin_prefetch(vp + 64L * 12288, 0, 3);
    }
    __builtin_amdgcn_s_wait_tensorcnt(0);
    __syncthreads();

    // ---- scores: S = Q(16x128) * K^T(128x64), 4 k-tiles x 4 K-chunks
    v8f sacc[4];
#pragma unroll
    for (int t = 0; t < 4; ++t) sacc[t] = {};
#pragma unroll
    for (int t = 0; t < 4; ++t) {
      const int krow = t * 16 + l;
#pragma unroll
      for (int kk = 0; kk < 4; ++kk) {
        v8bf blo = *(v8bf*)&sK[krow * 136 + kk * 32 + half * 8];
        v8bf bhi = *(v8bf*)&sK[krow * 136 + kk * 32 + 16 + half * 8];
        sacc[t] = wmma_bf16(qfrag[kk], cat16(blo, bhi), sacc[t]);
      }
    }

    // ---- scale + alibi + causal mask
    float sv[4][8];
#pragma unroll
    for (int t = 0; t < 4; ++t) {
      const int kpos = kb * 64 + t * 16 + l;
      const float kalibi = slope * (float)kpos;
#pragma unroll
      for (int r = 0; r < 8; ++r) {
        const int qr = qtile * 64 + wave * 16 + r + half * 8;
        float x = sacc[t][r] * INV_NORM + kalibi;
        sv[t][r] = (kpos <= qr) ? x : -3.0e38f;
      }
    }

    // ---- online softmax (reductions stay inside 16-lane halves: lanes 0-15
    //      carry row r, lanes 16-31 carry row r+8, matching the C layout)
#pragma unroll
    for (int r = 0; r < 8; ++r) {
      float mx = fmaxf(fmaxf(sv[0][r], sv[1][r]), fmaxf(sv[2][r], sv[3][r]));
#pragma unroll
      for (int off = 8; off > 0; off >>= 1) mx = fmaxf(mx, __shfl_xor(mx, off, 32));
      const float mnew  = fmaxf(mstate[r], mx);
      const float scale = __expf(mstate[r] - mnew);
      mstate[r] = mnew;
      float rsum = 0.0f;
#pragma unroll
      for (int t = 0; t < 4; ++t) {
        float p = __expf(sv[t][r] - mnew);
        sv[t][r] = p;
        rsum += p;
      }
#pragma unroll
      for (int off = 8; off > 0; off >>= 1) rsum += __shfl_xor(rsum, off, 32);
      lstate[r] = lstate[r] * scale + rsum;
#pragma unroll
      for (int td = 0; td < 8; ++td) ctxacc[td][r] *= scale;
    }

    // ---- P: C-layout regs -> row-major LDS (re-layout for A fragments)
#pragma unroll
    for (int t = 0; t < 4; ++t)
#pragma unroll
      for (int r = 0; r < 8; ++r)
        sP[wave][(r + half * 8) * 64 + t * 16 + l] = (bf16_t)sv[t][r];
    __syncthreads();

    // ---- ctx += P(16x64) * V(64x128): 2 K-chunks x 8 hd-tiles
#pragma unroll
    for (int kk = 0; kk < 2; ++kk) {
      v8bf plo = *(v8bf*)&sP[wave][l * 64 + kk * 32 + half * 8];
      v8bf phi = *(v8bf*)&sP[wave][l * 64 + kk * 32 + 16 + half * 8];
      v16bf pfrag = cat16(plo, phi);
#pragma unroll
      for (int td = 0; td < 8; ++td) {
        const int col = td * 16 + l;
        v8bf blo = *(v8bf*)&sVT[col * 72 + kk * 32 + half * 8];
        v8bf bhi = *(v8bf*)&sVT[col * 72 + kk * 32 + 16 + half * 8];
        ctxacc[td] = wmma_bf16(pfrag, cat16(blo, bhi), ctxacc[td]);
      }
    }
  }

  // ---- normalize and store ctx (merged head layout: col = h*128 + hd)
#pragma unroll
  for (int td = 0; td < 8; ++td) {
    const int col = h * HDIM + td * 16 + l;
#pragma unroll
    for (int r = 0; r < 8; ++r) {
      const long row = (long)b * SS + qtile * 64 + wave * 16 + r + half * 8;
      float v = ctxacc[td][r] / lstate[r];
      ctx[row * HH + col] = (bf16_t)v;
    }
  }
}

// ---------------------------------------------------------------- launcher
extern "C" void kernel_launch(void* const* d_in, const int* in_sizes, int n_in,
                              void* d_out, int out_size, void* d_ws, size_t ws_size,
                              hipStream_t stream) {
  const float* hs    = (const float*)d_in[0];  // hidden_states (2,1024,4096)
  const float* resid = (const float*)d_in[1];  // residual
  // d_in[2] alibi, d_in[3] attention_mask: recomputed analytically in-kernel
  const float* Wqkv  = (const float*)d_in[4];  // (4096, 12288)
  const float* bqkv  = (const float*)d_in[5];  // (12288,)
  const float* Wd    = (const float*)d_in[6];  // (4096, 4096)
  const float* bd    = (const float*)d_in[7];  // (4096,)
  float* out = (float*)d_out;

  char* ws = (char*)d_ws;
  bf16_t* hsb   = (bf16_t*)(ws);
  bf16_t* wqkvb = (bf16_t*)(ws + 16777216L);
  bf16_t* wdb   = (bf16_t*)(ws + 117440512L);
  bf16_t* qkvb  = (bf16_t*)(ws + 150994944L);
  bf16_t* ctxb  = (bf16_t*)(ws + 201326592L);

  bloom_cvt_f32_bf16<<<2048, 256, 0, stream>>>(hs, hsb, 8388608L);
  bloom_cvt_f32_bf16<<<8192, 256, 0, stream>>>(Wqkv, wqkvb, 50331648L);
  bloom_cvt_f32_bf16<<<4096, 256, 0, stream>>>(Wd, wdb, 16777216L);

  // fused QKV GEMM: [2048,4096] x [4096,12288] + b_qkv -> bf16
  bloom_gemm_bf16<false><<<dim3(96, 16), 256, 0, stream>>>(
      hsb, wqkvb, bqkv, nullptr, qkvb, nullptr, MTOK, 12288, HH);

  // flash attention over 64 head-batches x 16 q-tiles
  bloom_attn<<<dim3(16, 64), 128, 0, stream>>>(qkvb, ctxb);

  // dense projection: [2048,4096] x [4096,4096] + b_dense + residual -> f32
  bloom_gemm_bf16<true><<<dim3(32, 16), 256, 0, stream>>>(
      ctxb, wdb, bd, resid, nullptr, out, MTOK, HH, HH);
}
